// MultiHeadAttention_66855460929925
// MI455X (gfx1250) — compile-verified, matrix operand order checked by reading, not run on hardware
//
#include <hip/hip_runtime.h>
#include <hip/hip_bf16.h>
#include <math.h>

// ---------------------------------------------------------------------------
// MHA forward for MI455X (gfx1250, wave32, WMMA).
// D=1024, H=16, dk=64, B=2, T=2048, M = B*T = 4096.
// Compute-bound (~70 GFLOP vs ~50MB traffic) -> everything through
// v_wmma_f32_16x16x32_f16 (f16 in, f32 accumulate).
// Staging uses gfx1250 async global->LDS copies (ASYNCcnt).
// ---------------------------------------------------------------------------

typedef _Float16 half2v __attribute__((ext_vector_type(2)));
typedef _Float16 half8  __attribute__((ext_vector_type(8)));
typedef _Float16 half16 __attribute__((ext_vector_type(16)));
typedef float    float8 __attribute__((ext_vector_type(8)));

constexpr int D_MODEL = 1024;
constexpr int N_HEAD  = 16;
constexpr int DK      = 64;
constexpr int T_SEQ   = 2048;
constexpr int BATCH   = 2;
constexpr int M_TOT   = BATCH * T_SEQ;   // 4096
constexpr int KTILE   = 64;              // GEMM: K depth per LDS stage
constexpr int KEYB    = 64;              // attention: keys per block iter

static __device__ __forceinline__ half16 combine8(half8 lo, half8 hi) {
  half16 r;
#pragma unroll
  for (int i = 0; i < 8; ++i) { r[i] = lo[i]; r[i + 8] = hi[i]; }
  return r;
}

static __device__ __forceinline__ float8 wmma_f16(half16 a, half16 b, float8 c) {
  return __builtin_amdgcn_wmma_f32_16x16x32_f16(
      /*neg_a=*/false, a, /*neg_b=*/false, b,
      /*c_mod=*/(short)0, c, /*reuse_a=*/false, /*reuse_b=*/false);
}

// Async global -> LDS copy, 16 bytes per lane. ISA: GLOBAL_LOAD_ASYNC_TO_LDS_B128
// (VDST = LDS byte address VGPR, VADDR = 64-bit global address). Tracked by
// ASYNCcnt; wait with s_wait_asynccnt before the consuming barrier.
// LDS byte offset = low 32 bits of the flat address (ISA 10.2 aperture rule).
static __device__ __forceinline__ void async_cp_b128(void* lds, const void* gaddr) {
  unsigned int l = (unsigned int)(size_t)lds;
  asm volatile("global_load_async_to_lds_b128 %0, %1, off"
               :: "v"(l), "v"(gaddr) : "memory");
}
static __device__ __forceinline__ void async_wait0() {
  asm volatile("s_wait_asynccnt 0x0" ::: "memory");
}

// DPP cross-lane moves (ctrl must be a literal -> macro).
// Reduction over each 16-lane row: xor1, xor2, half-mirror, mirror.
#define DPP_F32(x, ctrl) \
  __int_as_float(__builtin_amdgcn_update_dpp( \
      0, __float_as_int(x), (ctrl), 0xF, 0xF, true))

static __device__ __forceinline__ float rowmax16(float x) {
  x = fmaxf(x, DPP_F32(x, 0xB1));   // quad_perm(1,0,3,2)  : xor 1
  x = fmaxf(x, DPP_F32(x, 0x4E));   // quad_perm(2,3,0,1)  : xor 2
  x = fmaxf(x, DPP_F32(x, 0x141));  // row_half_mirror     : pairs across quads
  x = fmaxf(x, DPP_F32(x, 0x140));  // row_mirror          : pairs across octs
  return x;
}
static __device__ __forceinline__ float rowsum16(float x) {
  x += DPP_F32(x, 0xB1);
  x += DPP_F32(x, 0x4E);
  x += DPP_F32(x, 0x141);
  x += DPP_F32(x, 0x140);
  return x;
}

// ---------------------------------------------------------------------------
// f32 -> f16 elementwise conversion
// ---------------------------------------------------------------------------
__global__ void cvt_f32_to_f16(const float* __restrict__ src,
                               _Float16* __restrict__ dst, int n) {
  int i = blockIdx.x * blockDim.x + threadIdx.x;
  if (i < n) dst[i] = (_Float16)src[i];
}

// ---------------------------------------------------------------------------
// NT GEMM: Y[m,n] = sum_k A[m,k]*W[n,k] + bias[n]
//   A: (4096, 1024) f16 row-major,  W: (1024, 1024) f16 row-major.
// LAYOUT==0: write f16 to (B, H, T, DK) tensor `outh`  (QKV projections)
// LAYOUT==1: write f32 to (M, 1024) row-major `outf`   (final output proj)
// Block: 256 threads (8 waves). Block tile 128x128, wave tile 32x64.
// K tile 64 deep -> 16 WMMAs per barrier pair. Staging via async-to-LDS.
// ---------------------------------------------------------------------------
template <int LAYOUT>
__global__ __launch_bounds__(256)
void gemm_nt_wmma(const _Float16* __restrict__ A,
                  const _Float16* __restrict__ W,
                  const float* __restrict__ bias,
                  _Float16* __restrict__ outh,
                  float* __restrict__ outf) {
  __shared__ _Float16 As[128 * KTILE];   // 16 KB
  __shared__ _Float16 Bs[128 * KTILE];   // 16 KB

  const int tid    = threadIdx.x;
  const int wave   = tid >> 5;
  const int lane   = tid & 31;
  const int lane15 = lane & 15;
  const int hi16   = (lane >= 16) ? 1 : 0;
  const int kbase  = hi16 ? 8 : 0;

  const int wm = wave & 3;        // 0..3  -> m offset wm*32
  const int wn = wave >> 2;       // 0..1  -> n offset wn*64
  const int m0 = blockIdx.y * 128;
  const int n0 = blockIdx.x * 128;

  float8 acc[2][4];
#pragma unroll
  for (int mi = 0; mi < 2; ++mi)
#pragma unroll
    for (int ni = 0; ni < 4; ++ni) acc[mi][ni] = (float8)(0.0f);

  for (int kb = 0; kb < D_MODEL; kb += KTILE) {
    __syncthreads();
    // cooperative stage via async global->LDS: 128 rows x 64 halfs each
#pragma unroll
    for (int rep = 0; rep < 4; ++rep) {
      int seg = tid + rep * 256;          // 0..1023
      int row = seg >> 3;                 // 0..127
      int ks  = (seg & 7) * 8;            // 0..56
      async_cp_b128(&As[row * KTILE + ks],
                    &A[(size_t)(m0 + row) * D_MODEL + kb + ks]);
      async_cp_b128(&Bs[row * KTILE + ks],
                    &W[(size_t)(n0 + row) * D_MODEL + kb + ks]);
      if (rep == 0 && kb + KTILE < D_MODEL) {
        // hint next K tile into cache (global_prefetch_b8)
        __builtin_prefetch(&A[(size_t)(m0 + row) * D_MODEL + kb + KTILE + ks], 0, 0);
        __builtin_prefetch(&W[(size_t)(n0 + row) * D_MODEL + kb + KTILE + ks], 0, 0);
      }
    }
    async_wait0();
    __syncthreads();

#pragma unroll
    for (int ks2 = 0; ks2 < 2; ++ks2) {   // two 32-deep WMMA k-steps
      const int ko = ks2 * 32 + kbase;
      half16 af[2], bf[4];
#pragma unroll
      for (int mi = 0; mi < 2; ++mi) {
        int row = wm * 32 + mi * 16 + lane15;
        af[mi] = combine8(*(const half8*)&As[row * KTILE + ko],
                          *(const half8*)&As[row * KTILE + ko + 16]);
      }
#pragma unroll
      for (int ni = 0; ni < 4; ++ni) {
        int row = wn * 64 + ni * 16 + lane15;
        bf[ni] = combine8(*(const half8*)&Bs[row * KTILE + ko],
                          *(const half8*)&Bs[row * KTILE + ko + 16]);
      }
#pragma unroll
      for (int mi = 0; mi < 2; ++mi)
#pragma unroll
        for (int ni = 0; ni < 4; ++ni)
          acc[mi][ni] = wmma_f16(af[mi], bf[ni], acc[mi][ni]);
    }
  }

  // writeback (C layout: VGPR r -> row r (lanes 0-15) / row r+8 (lanes 16-31))
#pragma unroll
  for (int ni = 0; ni < 4; ++ni) {
    const int gn = n0 + wn * 64 + ni * 16 + lane15;
    const float bv = bias[gn];
#pragma unroll
    for (int mi = 0; mi < 2; ++mi) {
#pragma unroll
      for (int r = 0; r < 8; ++r) {
        int gm = m0 + wm * 32 + mi * 16 + r + hi16 * 8;
        float v = acc[mi][ni][r] + bv;
        if constexpr (LAYOUT == 0) {
          int b = gm >> 11, t = gm & (T_SEQ - 1);
          int h = gn >> 6,  d = gn & (DK - 1);
          outh[(((size_t)(b * N_HEAD + h) * T_SEQ) + t) * DK + d] = (_Float16)v;
        } else {
          outf[(size_t)gm * D_MODEL + gn] = v;
        }
      }
    }
  }
}

// ---------------------------------------------------------------------------
// Flash attention: per block (128 threads / 4 waves): one (b,h), 64 query rows.
// Each wave owns 16 queries. 64 keys per iteration:
//   - K staged into LDS via async-to-LDS; V staged transposed (b32 pair packing)
//   - scores: 8 WMMAs (4 key tiles x dk=64 in 2 k-steps)
//   - online softmax, row stats reduced with DPP (no LDS bpermute)
//   - P -> per-wave LDS -> A fragments; PV: 8 WMMAs
// Q pre-scaled by 1/sqrt(dk) at fragment load.
// ---------------------------------------------------------------------------
__global__ __launch_bounds__(128)
void attn_wmma(const _Float16* __restrict__ Qh,
               const _Float16* __restrict__ Kh,
               const _Float16* __restrict__ Vh,
               _Float16* __restrict__ Ch) {
  __shared__ _Float16 Ksh[KEYB * 64];        // 8 KB (key-major, straight)
  __shared__ _Float16 Vt[64 * KEYB];         // 8 KB (dim-major = transposed)
  __shared__ _Float16 Pws[4 * 16 * KEYB];    // 8 KB (per-wave P scratch)

  const int tid    = threadIdx.x;
  const int wave   = tid >> 5;
  const int lane   = tid & 31;
  const int lane15 = lane & 15;
  const int hi16   = (lane >= 16) ? 1 : 0;
  const int kbase  = hi16 ? 8 : 0;

  const int bh = blockIdx.y;               // 0..31
  const int b  = bh >> 4;
  const int h  = bh & 15;
  const int q0 = blockIdx.x * 64;

  const size_t bh_off = (size_t)bh * T_SEQ * DK;
  const _Float16* Qb = Qh + bh_off;
  const _Float16* Kb = Kh + bh_off;
  const _Float16* Vb = Vh + bh_off;

  // Q fragments for this wave's 16 rows, pre-scaled by 1/sqrt(64) = 0.125
  const int qrow = q0 + wave * 16 + lane15;
  half16 qf[2];
#pragma unroll
  for (int kd = 0; kd < 2; ++kd) {
    half16 q = combine8(
        *(const half8*)&Qb[(size_t)qrow * DK + kd * 32 + kbase],
        *(const half8*)&Qb[(size_t)qrow * DK + kd * 32 + kbase + 16]);
#pragma unroll
    for (int i = 0; i < 16; ++i) q[i] = q[i] * (_Float16)0.125f;
    qf[kd] = q;
  }

  float8 o[4];
#pragma unroll
  for (int j = 0; j < 4; ++j) o[j] = (float8)(0.0f);
  float mrow[8], lrow[8];
#pragma unroll
  for (int r = 0; r < 8; ++r) { mrow[r] = -__builtin_inff(); lrow[r] = 0.0f; }

  for (int kk = 0; kk < T_SEQ; kk += KEYB) {
    __syncthreads();
    // stage K straight via async-to-LDS: 64 keys x 64 dims
#pragma unroll
    for (int rep = 0; rep < 2; ++rep) {
      int seg = tid + rep * 128;      // 0..255
      int row = seg >> 2;             // 0..63
      int db  = (seg & 3) * 16;       // 0,16,32,48
      async_cp_b128(&Ksh[row * 64 + db],
                    &Kb[(size_t)(kk + row) * DK + db]);
      async_cp_b128(&Ksh[row * 64 + db + 8],
                    &Kb[(size_t)(kk + row) * DK + db + 8]);
    }
    // stage V transposed: pack key pairs -> b32 LDS stores
#pragma unroll
    for (int rep = 0; rep < 2; ++rep) {
      int idx = tid + rep * 128;      // 0..255
      int kp  = idx >> 3;             // 0..31 -> keys 2kp, 2kp+1
      int db  = (idx & 7) * 8;        // 0..56
      half8 va = *(const half8*)&Vb[(size_t)(kk + 2 * kp) * DK + db];
      half8 vb = *(const half8*)&Vb[(size_t)(kk + 2 * kp + 1) * DK + db];
#pragma unroll
      for (int i = 0; i < 8; ++i) {
        half2v p; p[0] = va[i]; p[1] = vb[i];
        *(half2v*)&Vt[(db + i) * KEYB + 2 * kp] = p;
      }
    }
    async_wait0();
    __syncthreads();

    // scores for 64 keys = four 16x16 tiles
    float8 s[4];
#pragma unroll
    for (int nt = 0; nt < 4; ++nt) {
      float8 sc = (float8)(0.0f);
      int key = nt * 16 + lane15;
#pragma unroll
      for (int kd = 0; kd < 2; ++kd) {
        half16 bf = combine8(*(const half8*)&Ksh[key * 64 + kd * 32 + kbase],
                             *(const half8*)&Ksh[key * 64 + kd * 32 + kbase + 16]);
        sc = wmma_f16(qf[kd], bf, sc);
      }
      s[nt] = sc;
    }

    // online softmax (row stats; DPP reductions over 16-lane rows)
    float sf[8];
#pragma unroll
    for (int r = 0; r < 8; ++r) {
      float loc = fmaxf(fmaxf(s[0][r], s[1][r]), fmaxf(s[2][r], s[3][r]));
      loc = rowmax16(loc);
      float mn = fmaxf(mrow[r], loc);
      sf[r] = __expf(mrow[r] - mn);
      mrow[r] = mn;
      float rs = 0.0f;
#pragma unroll
      for (int nt = 0; nt < 4; ++nt) {
        float p = __expf(s[nt][r] - mn);
        s[nt][r] = p;
        rs += p;
      }
      rs = rowsum16(rs);
      lrow[r] = lrow[r] * sf[r] + rs;
    }
#pragma unroll
    for (int j = 0; j < 4; ++j)
#pragma unroll
      for (int r = 0; r < 8; ++r) o[j][r] = o[j][r] * sf[r];

    // P: C-layout -> LDS -> A-fragment layout (per-wave region)
    _Float16* Pw = &Pws[wave * 16 * KEYB];
#pragma unroll
    for (int r = 0; r < 8; ++r) {
      int prow = r + hi16 * 8;
#pragma unroll
      for (int nt = 0; nt < 4; ++nt)
        Pw[prow * KEYB + nt * 16 + lane15] = (_Float16)s[nt][r];
    }
    __syncthreads();

    half16 pf[2];
#pragma unroll
    for (int kd2 = 0; kd2 < 2; ++kd2)
      pf[kd2] = combine8(
          *(const half8*)&Pw[lane15 * KEYB + kd2 * 32 + kbase],
          *(const half8*)&Pw[lane15 * KEYB + kd2 * 32 + kbase + 16]);
#pragma unroll
    for (int j = 0; j < 4; ++j) {
      int dc = j * 16 + lane15;
#pragma unroll
      for (int kd2 = 0; kd2 < 2; ++kd2) {
        half16 vf = combine8(
            *(const half8*)&Vt[dc * KEYB + kd2 * 32 + kbase],
            *(const half8*)&Vt[dc * KEYB + kd2 * 32 + kbase + 16]);
        o[j] = wmma_f16(pf[kd2], vf, o[j]);
      }
    }
  }

  // normalize and write context in (B, T, D) f16
#pragma unroll
  for (int j = 0; j < 4; ++j) {
#pragma unroll
    for (int r = 0; r < 8; ++r) {
      int row = q0 + wave * 16 + r + hi16 * 8;
      float val = o[j][r] / lrow[r];
      Ch[((size_t)(b * T_SEQ + row)) * D_MODEL + h * DK + j * 16 + lane15] =
          (_Float16)val;
    }
  }
}

// ---------------------------------------------------------------------------
// Launch
// ---------------------------------------------------------------------------
extern "C" void kernel_launch(void* const* d_in, const int* in_sizes, int n_in,
                              void* d_out, int out_size, void* d_ws, size_t ws_size,
                              hipStream_t stream) {
  const float* x  = (const float*)d_in[0];
  const float* Wq = (const float*)d_in[1];
  const float* bq = (const float*)d_in[2];
  const float* Wk = (const float*)d_in[3];
  const float* bk = (const float*)d_in[4];
  const float* Wv = (const float*)d_in[5];
  const float* bv = (const float*)d_in[6];
  const float* Wo = (const float*)d_in[7];
  const float* bo = (const float*)d_in[8];

  char* ws = (char*)d_ws;
  const size_t MB = 1024 * 1024;
  _Float16* xh  = (_Float16*)(ws + 0 * MB);    // 8 MB  (4096x1024)
  _Float16* Wqh = (_Float16*)(ws + 8 * MB);    // 2 MB
  _Float16* Wkh = (_Float16*)(ws + 10 * MB);   // 2 MB
  _Float16* Wvh = (_Float16*)(ws + 12 * MB);   // 2 MB
  _Float16* Woh = (_Float16*)(ws + 14 * MB);   // 2 MB
  _Float16* Qh  = (_Float16*)(ws + 16 * MB);   // 8 MB  (B,H,T,DK)
  _Float16* Kh  = (_Float16*)(ws + 24 * MB);   // 8 MB
  _Float16* Vh  = (_Float16*)(ws + 32 * MB);   // 8 MB
  _Float16* Ch  = (_Float16*)(ws + 40 * MB);   // 8 MB  (B,T,D)

  const int nX = M_TOT * D_MODEL;       // 4194304
  const int nW = D_MODEL * D_MODEL;     // 1048576
  cvt_f32_to_f16<<<(nX + 255) / 256, 256, 0, stream>>>(x, xh, nX);
  cvt_f32_to_f16<<<(nW + 255) / 256, 256, 0, stream>>>(Wq, Wqh, nW);
  cvt_f32_to_f16<<<(nW + 255) / 256, 256, 0, stream>>>(Wk, Wkh, nW);
  cvt_f32_to_f16<<<(nW + 255) / 256, 256, 0, stream>>>(Wv, Wvh, nW);
  cvt_f32_to_f16<<<(nW + 255) / 256, 256, 0, stream>>>(Wo, Woh, nW);

  dim3 ggrid(D_MODEL / 128, M_TOT / 128);   // (8, 32)
  gemm_nt_wmma<0><<<ggrid, 256, 0, stream>>>(xh, Wqh, bq, Qh, nullptr);
  gemm_nt_wmma<0><<<ggrid, 256, 0, stream>>>(xh, Wkh, bk, Kh, nullptr);
  gemm_nt_wmma<0><<<ggrid, 256, 0, stream>>>(xh, Wvh, bv, Vh, nullptr);

  dim3 agrid(T_SEQ / 64, BATCH * N_HEAD);   // (32, 32)
  attn_wmma<<<agrid, 128, 0, stream>>>(Qh, Kh, Vh, Ch);

  gemm_nt_wmma<1><<<ggrid, 256, 0, stream>>>(Ch, Woh, bo, nullptr, (float*)d_out);
}
